// MultiHeadSelfAttention_66683662238493
// MI455X (gfx1250) — matrix operand order repaired and checked by reading, not compiled, not run on hardware
//
#include <hip/hip_runtime.h>

// ---------------------------------------------------------------------------
// MultiHeadSelfAttention for MI455X (gfx1250, wave32, WMMA).
// bf16 WMMA (v_wmma_f32_16x16x32_bf16) with fp32 accumulation, double-buffered
// LDS pipelined via GLOBAL_LOAD_ASYNC_TO_LDS (ASYNCcnt) and a TDM
// TENSOR_LOAD_TO_LDS (TENSORcnt) path for the attention K tiles.
// ---------------------------------------------------------------------------

#define BATCH 4
#define SEQ   2048
#define DIM   1024
#define NHEAD 16
#define HDIM  64

typedef __attribute__((ext_vector_type(16))) __bf16 v16bf;
typedef __attribute__((ext_vector_type(8)))  float  v8f;
typedef __attribute__((ext_vector_type(4)))  unsigned u32x4;
typedef __attribute__((ext_vector_type(2)))  unsigned u32x2;
typedef __attribute__((ext_vector_type(4)))  float  f32x4;

// GCC-style vectors to match clang builtin signatures exactly.
typedef int      gv4i  __attribute__((vector_size(16)));
typedef unsigned gv4u  __attribute__((vector_size(16)));
typedef int      gv8i  __attribute__((vector_size(32)));

#if defined(__HIP_DEVICE_COMPILE__) && \
    __has_builtin(__builtin_amdgcn_global_load_async_to_lds_b128)
#define HAVE_ASYNC 1
#else
#define HAVE_ASYNC 0
#endif

#if defined(__HIP_DEVICE_COMPILE__) && \
    __has_builtin(__builtin_amdgcn_tensor_load_to_lds) && \
    __has_builtin(__builtin_amdgcn_s_wait_tensorcnt)
#define HAVE_TDM 1
#else
#define HAVE_TDM 0
#endif

union Frag { v16bf v; unsigned u[8]; };

__device__ __forceinline__ unsigned short f2bf(float x) {
  union { float f; unsigned u; } v; v.f = x;
  unsigned r = v.u + 0x7FFFu + ((v.u >> 16) & 1u);   // round-to-nearest-even
  return (unsigned short)(r >> 16);
}
__device__ __forceinline__ float bf2f(unsigned short h) {
  union { float f; unsigned u; } v; v.u = ((unsigned)h) << 16;
  return v.f;
}

#if HAVE_ASYNC
__device__ __forceinline__ void async_copy_b128(const void* gsrc, void* ldst) {
  __builtin_amdgcn_global_load_async_to_lds_b128(
      (__attribute__((address_space(1))) gv4i*)(uintptr_t)gsrc,
      (__attribute__((address_space(3))) gv4i*)(unsigned)(uintptr_t)ldst,
      0, 0);
}
#endif

__device__ __forceinline__ void wait_async0() {
#if HAVE_ASYNC
#if __has_builtin(__builtin_amdgcn_s_wait_asynccnt)
  __builtin_amdgcn_s_wait_asynccnt(0);
#else
  asm volatile("s_wait_asynccnt 0" ::: "memory");
#endif
#endif
}

#if HAVE_TDM
// 2D TDM tile load: bf16 elements, optional LDS row padding.
// padi: pad_interval code (interval = 2<<padi dwords); pada: pad_amount-1 dwords.
__device__ __forceinline__ void tdm_load_2d(unsigned lds_byte, const void* g,
                                            unsigned tile0, unsigned tile1,
                                            unsigned ten0, unsigned ten1,
                                            unsigned stride0,
                                            unsigned padi, unsigned pada) {
  unsigned long long ga = (unsigned long long)(uintptr_t)g;
  gv4u g0;
  g0[0] = 1u;                                            // count=1, user D#
  g0[1] = lds_byte;                                      // lds_addr
  g0[2] = (unsigned)ga;                                  // global_addr[31:0]
  g0[3] = (unsigned)((ga >> 32) & 0x01ffffffu) | (2u << 30); // [56:32] | type=2
  gv8i g1;
  g1[0] = (int)((1u << 16) | (1u << 20) | (padi << 22) | (pada << 25));
  g1[1] = (int)((ten0 & 0xffffu) << 16);                 // tensor_dim0[15:0]
  g1[2] = (int)((ten0 >> 16) | ((ten1 & 0xffffu) << 16));
  g1[3] = (int)((ten1 >> 16) | (tile0 << 16));           // tile_dim0
  g1[4] = (int)(tile1 & 0xffffu);                        // tile_dim1, tile_dim2=0
  g1[5] = (int)stride0;                                  // tensor_dim0_stride
  g1[6] = 0;
  g1[7] = 0;
  gv4i gz = {0, 0, 0, 0};
#if __clang_major__ >= 23
  gv8i gz8 = {0, 0, 0, 0, 0, 0, 0, 0};
  __builtin_amdgcn_tensor_load_to_lds(g0, g1, gz, gz, gz8, 0);
#else
  __builtin_amdgcn_tensor_load_to_lds(g0, g1, gz, gz, 0);
#endif
}
#endif

// A-matrix fragment (16x32 bf16) from row-major [M][stride] LDS.
__device__ __forceinline__ v16bf frag_a(const unsigned short* base, int stride) {
  const int lane = threadIdx.x & 31;
  const int m = lane & 15, h = lane >> 4;
  const unsigned* p0 = (const unsigned*)(base + m * stride + 16 * h);
  const unsigned* p1 = (const unsigned*)(base + m * stride + 16 * h + 8);
  Frag f;
  f.u[0] = p0[0]; f.u[1] = p0[1]; f.u[2] = p0[2]; f.u[3] = p0[3];
  f.u[4] = p1[0]; f.u[5] = p1[1]; f.u[6] = p1[2]; f.u[7] = p1[3];
  return f.v;
}

// B-matrix fragment (32x16 bf16) from N-major [N][stride] LDS.
__device__ __forceinline__ v16bf frag_b(const unsigned short* base, int stride) {
  const int lane = threadIdx.x & 31;
  const int n = lane & 15, h = lane >> 4;
  const unsigned* p = (const unsigned*)(base + n * stride + 16 * h);
  Frag f;
#pragma unroll
  for (int i = 0; i < 8; i++) f.u[i] = p[i];
  return f.v;
}

__device__ __forceinline__ v8f wmma_bf16(v16bf a, v16bf b, v8f c) {
  return __builtin_amdgcn_wmma_f32_16x16x32_bf16(false, a, false, b,
                                                 (short)0, c, false, false);
}

__device__ __forceinline__ float redmax16(float v) {
  v = fmaxf(v, __shfl_xor(v, 1, 32));
  v = fmaxf(v, __shfl_xor(v, 2, 32));
  v = fmaxf(v, __shfl_xor(v, 4, 32));
  v = fmaxf(v, __shfl_xor(v, 8, 32));
  return v;
}
__device__ __forceinline__ float redsum16(float v) {
  v += __shfl_xor(v, 1, 32);
  v += __shfl_xor(v, 2, 32);
  v += __shfl_xor(v, 4, 32);
  v += __shfl_xor(v, 8, 32);
  return v;
}

// ---------------------------------------------------------------------------
__global__ __launch_bounds__(256) void cvt_bf16(const float* __restrict__ in,
                                                unsigned short* __restrict__ out,
                                                int n) {
  int i = (blockIdx.x * 256 + threadIdx.x) * 4;
  if (i >= n) return;
  f32x4 v = *(const f32x4*)(in + i);
  u32x2 o;
  o[0] = (unsigned)f2bf(v[0]) | ((unsigned)f2bf(v[1]) << 16);
  o[1] = (unsigned)f2bf(v[2]) | ((unsigned)f2bf(v[3]) << 16);
  *(u32x2*)(out + i) = o;
}

// ---------------------------------------------------------------------------
// QKV GEMM (x[8192,1024] @ Wqkv[1024,3072]) + bias + RoPE + head-major scatter.
// 128x128x32 tiles, double-buffered LDS, async-to-LDS pipeline.
// ---------------------------------------------------------------------------
__global__ __launch_bounds__(256) void qkv_gemm_rope(
    const unsigned short* __restrict__ xb,     // [8192][1024] bf16
    const unsigned short* __restrict__ wb,     // [1024][3072] bf16
    const float* __restrict__ bqkv,            // [3072]
    unsigned short* __restrict__ qb,           // [B*NH][S][HD] (pre-scaled)
    unsigned short* __restrict__ kb,
    unsigned short* __restrict__ vb)
{
  __shared__ unsigned short smem[20480];       // 40 KB: A0,A1,B0,B1 / Cst
  unsigned short* Cst = smem;                  // [128][136] bf16 (epilogue)

  const int tid = threadIdx.x;
  const int wave = tid >> 5;
  const int lane = tid & 31;
  const int hf = lane >> 4, ln = lane & 15;
  const int wm = wave & 3, wn = wave >> 2;
  const int rowbase = blockIdx.y * 128;
  const int colbase = blockIdx.x * 128;

  const v8f vzero = {0.f, 0.f, 0.f, 0.f, 0.f, 0.f, 0.f, 0.f};
  v8f acc[2][4];
#pragma unroll
  for (int i = 0; i < 2; i++)
#pragma unroll
    for (int j = 0; j < 4; j++) acc[i][j] = vzero;

  auto load_tile = [&](int kt, unsigned short* A, unsigned short* B) {
    // A tile 128x32 row-major
#pragma unroll
    for (int i = 0; i < 2; i++) {
      int c = tid + 256 * i;
      int r = c >> 2, col = (c & 3) * 8;
      const unsigned short* src = xb + (size_t)(rowbase + r) * DIM + kt + col;
      unsigned short* dst = A + r * 40 + col;
#if HAVE_ASYNC
      async_copy_b128(src, dst);
#else
      *(u32x4*)dst = *(const u32x4*)src;
#endif
    }
    // B tile 32x128, transposed into [n][k] with dword-combined stores
    {
      int kp = (tid & 15) * 2;
      int nc = (tid >> 4) * 8;
      const unsigned short* s0 = wb + (size_t)(kt + kp) * (3 * DIM) + colbase + nc;
      u32x4 d0 = *(const u32x4*)s0;
      u32x4 d1 = *(const u32x4*)(s0 + 3 * DIM);
#pragma unroll
      for (int e = 0; e < 8; e++) {
        unsigned lo = (d0[e >> 1] >> (16 * (e & 1))) & 0xffffu;
        unsigned hi = (d1[e >> 1] >> (16 * (e & 1))) & 0xffffu;
        *(unsigned*)(B + (nc + e) * 40 + kp) = lo | (hi << 16);
      }
    }
  };

  load_tile(0, smem, smem + 10240);
  wait_async0();
  __syncthreads();

  for (int t = 0; t < DIM / 32; ++t) {
    unsigned short* Ac = smem + (t & 1) * 5120;
    unsigned short* Bc = smem + 10240 + (t & 1) * 5120;
    if (t + 1 < DIM / 32)
      load_tile((t + 1) * 32, smem + ((t + 1) & 1) * 5120,
                smem + 10240 + ((t + 1) & 1) * 5120);

    v16bf af[2], bfm[4];
#pragma unroll
    for (int i = 0; i < 2; i++) af[i] = frag_a(Ac + (32 * wm + 16 * i) * 40, 40);
#pragma unroll
    for (int j = 0; j < 4; j++) bfm[j] = frag_b(Bc + (64 * wn + 16 * j) * 40, 40);
#pragma unroll
    for (int i = 0; i < 2; i++)
#pragma unroll
      for (int j = 0; j < 4; j++) acc[i][j] = wmma_bf16(af[i], bfm[j], acc[i][j]);

    wait_async0();
    __syncthreads();
  }

  // Stage C tile (bf16) so RoPE can pair (d, d+32) across lanes/waves.
#pragma unroll
  for (int i = 0; i < 2; i++)
#pragma unroll
    for (int j = 0; j < 4; j++)
#pragma unroll
      for (int r = 0; r < 8; r++)
        Cst[(32 * wm + 16 * i + r + 8 * hf) * 136 + 64 * wn + 16 * j + ln] =
            f2bf(acc[i][j][r]);
  __syncthreads();

  const float L2I = 0.41524101186092029f;      // log2(10000)/32
  for (int i = 0; i < 64; i++) {
    int e = i * 256 + tid;
    int m = e >> 7, n = e & 127;
    int col = colbase + n;
    int which = col >> 10;
    int hcol = col & 1023;
    int head = hcol >> 6;
    int d = hcol & 63;
    int rowg = rowbase + m;
    int b = rowg >> 11;
    int spos = rowg & 2047;
    float v0 = bf2f(Cst[m * 136 + n]) + bqkv[col];
    float outv;
    unsigned short* dst;
    if (which == 2) {
      outv = v0;
      dst = vb;
    } else {
      int d0 = (d < 32) ? d : d - 32;
      float invf = exp2f(-(float)d0 * L2I);
      float ang = (float)spos * invf;
      float s, c;
      __sincosf(ang, &s, &c);
      if (d < 32) {
        float v2 = bf2f(Cst[m * 136 + n + 32]) + bqkv[col + 32];
        outv = v0 * c - v2 * s;
      } else {
        float v1 = bf2f(Cst[m * 136 + n - 32]) + bqkv[col - 32];
        outv = v1 * s + v0 * c;
      }
      if (which == 0) outv *= 0.125f;          // fold 1/sqrt(HD) into Q
      dst = (which == 0) ? qb : kb;
    }
    size_t oi = (((size_t)(b * NHEAD + head)) * SEQ + spos) * HDIM + d;
    dst[oi] = f2bf(outv);
  }
}

// ---------------------------------------------------------------------------
// Flash attention: 128 query rows x one (b,h) per block; KV tiles of 64.
// K tiles arrive via TDM (padded into LDS by the descriptor) when available.
// ---------------------------------------------------------------------------
__global__ __launch_bounds__(256) void attn_kernel(
    const unsigned short* __restrict__ qbuf,
    const unsigned short* __restrict__ kbuf,
    const unsigned short* __restrict__ vbuf,
    unsigned short* __restrict__ obuf)         // [B*S][DIM] bf16
{
  __shared__ unsigned short smem[27648];       // 55.3 KB
  unsigned short* Qlds = smem;                 // [128][72]
  unsigned short* Klds = smem + 9216;          // [64][72]  N-major for QK^T
  unsigned short* Vt   = smem + 13824;         // [64][72]  Vt[d][kv]
  unsigned short* Plds = smem + 18432;         // [128][72]

  const int tid = threadIdx.x;
  const int wave = tid >> 5;
  const int lane = tid & 31;
  const int hf = lane >> 4, ln = lane & 15;
  const int bh = blockIdx.y;
  const int m0 = blockIdx.x * 128;
  const size_t base = (size_t)bh * SEQ * HDIM;

  // Q tile 128x64 -> LDS
#pragma unroll
  for (int i = 0; i < 4; i++) {
    int c = tid + 256 * i;
    int r = c >> 3, col = (c & 7) * 8;
    const unsigned short* src = qbuf + base + (size_t)(m0 + r) * HDIM + col;
    unsigned short* dst = Qlds + r * 72 + col;
#if HAVE_ASYNC
    async_copy_b128(src, dst);
#else
    *(u32x4*)dst = *(const u32x4*)src;
#endif
  }
  wait_async0();
  __syncthreads();

  v16bf qa[2];
  qa[0] = frag_a(Qlds + (16 * wave) * 72 + 0, 72);
  qa[1] = frag_a(Qlds + (16 * wave) * 72 + 32, 72);

  const v8f vzero = {0.f, 0.f, 0.f, 0.f, 0.f, 0.f, 0.f, 0.f};
  v8f o[4];
#pragma unroll
  for (int j = 0; j < 4; j++) o[j] = vzero;
  float mi[8], li[8];
#pragma unroll
  for (int r = 0; r < 8; r++) { mi[r] = -3.0e38f; li[r] = 0.f; }

  for (int kv = 0; kv < SEQ; kv += 64) {
    // K tile 64x64 -> Klds (row stride 72 via TDM pad: 32 dwords + 4 pad)
#if HAVE_TDM
    if (wave == 0)
      tdm_load_2d((unsigned)(uintptr_t)(void*)Klds,
                  kbuf + base + (size_t)kv * HDIM,
                  /*tile*/ 64, 64, /*tensor*/ 64, 64, /*stride0*/ 64,
                  /*pad_interval=32dw*/ 4u, /*pad=4dw*/ 3u);
#else
#pragma unroll
    for (int i = 0; i < 2; i++) {
      int c = tid + 256 * i;
      int r = c >> 3, col = (c & 7) * 8;
      const unsigned short* src = kbuf + base + (size_t)(kv + r) * HDIM + col;
      unsigned short* dst = Klds + r * 72 + col;
#if HAVE_ASYNC
      async_copy_b128(src, dst);
#else
      *(u32x4*)dst = *(const u32x4*)src;
#endif
    }
#endif
    // V tile 64x64 transposed into Vt[d][kv], dword-combined stores
    {
      int kvp = (tid & 31) * 2;
      int nc = (tid >> 5) * 8;
      const unsigned short* s0 = vbuf + base + (size_t)(kv + kvp) * HDIM + nc;
      u32x4 d0 = *(const u32x4*)s0;
      u32x4 d1 = *(const u32x4*)(s0 + HDIM);
#pragma unroll
      for (int e = 0; e < 8; e++) {
        unsigned lo = (d0[e >> 1] >> (16 * (e & 1))) & 0xffffu;
        unsigned hi = (d1[e >> 1] >> (16 * (e & 1))) & 0xffffu;
        *(unsigned*)(Vt + (nc + e) * 72 + kvp) = lo | (hi << 16);
      }
    }
#if HAVE_TDM
    if (wave == 0) __builtin_amdgcn_s_wait_tensorcnt(0);
#endif
    wait_async0();
    __syncthreads();

    // S = Q @ K^T  (Q pre-scaled by 1/sqrt(HD))
    v8f s[4];
#pragma unroll
    for (int j = 0; j < 4; j++) {
      s[j] = vzero;
#pragma unroll
      for (int kk = 0; kk < 2; kk++) {
        v16bf bk = frag_b(Klds + (16 * j) * 72 + 32 * kk, 72);
        s[j] = wmma_bf16(qa[kk], bk, s[j]);
      }
    }

    // Online softmax per row (r + 8*hf within this wave's 16 rows)
#pragma unroll
    for (int r = 0; r < 8; r++) {
      float v = fmaxf(fmaxf(s[0][r], s[1][r]), fmaxf(s[2][r], s[3][r]));
      v = redmax16(v);
      float mnew = fmaxf(mi[r], v);
      float alpha = __expf(mi[r] - mnew);
      float rsum = 0.f;
#pragma unroll
      for (int j = 0; j < 4; j++) {
        float p = __expf(s[j][r] - mnew);
        s[j][r] = p;
        rsum += p;
      }
      rsum = redsum16(rsum);
      li[r] = li[r] * alpha + rsum;
      mi[r] = mnew;
#pragma unroll
      for (int j = 0; j < 4; j++) o[j][r] = o[j][r] * alpha;
    }

    // P -> LDS (wave-private rows; same-wave DS ops stay in order)
#pragma unroll
    for (int j = 0; j < 4; j++)
#pragma unroll
      for (int r = 0; r < 8; r++)
        Plds[(16 * wave + r + 8 * hf) * 72 + 16 * j + ln] = f2bf(s[j][r]);

    // O += P @ V
#pragma unroll
    for (int kk = 0; kk < 2; kk++) {
      v16bf pa = frag_a(Plds + (16 * wave) * 72 + 32 * kk, 72);
#pragma unroll
      for (int j = 0; j < 4; j++) {
        v16bf bv = frag_b(Vt + (16 * j) * 72 + 32 * kk, 72);
        o[j] = wmma_bf16(pa, bv, o[j]);
      }
    }
    __syncthreads();
  }

  const int b = bh >> 4, h = bh & 15;
#pragma unroll
  for (int j = 0; j < 4; j++)
#pragma unroll
    for (int r = 0; r < 8; r++) {
      int srow = m0 + 16 * wave + r + 8 * hf;
      float val = o[j][r] / li[r];
      obuf[((size_t)(b * SEQ + srow)) * DIM + h * HDIM + 16 * j + ln] = f2bf(val);
    }
}

// ---------------------------------------------------------------------------
// Output projection: attn[8192,1024] @ Wout[1024,1024] + bout -> fp32 out
// ---------------------------------------------------------------------------
__global__ __launch_bounds__(256) void out_gemm(
    const unsigned short* __restrict__ ab,
    const unsigned short* __restrict__ wb,
    const float* __restrict__ bout,
    float* __restrict__ out)
{
  __shared__ unsigned short smem[20480];       // A0,A1,B0,B1

  const int tid = threadIdx.x;
  const int wave = tid >> 5;
  const int lane = tid & 31;
  const int hf = lane >> 4, ln = lane & 15;
  const int wm = wave & 3, wn = wave >> 2;
  const int rowbase = blockIdx.y * 128;
  const int colbase = blockIdx.x * 128;

  const v8f vzero = {0.f, 0.f, 0.f, 0.f, 0.f, 0.f, 0.f, 0.f};
  v8f acc[2][4];
#pragma unroll
  for (int i = 0; i < 2; i++)
#pragma unroll
    for (int j = 0; j < 4; j++) acc[i][j] = vzero;

  auto load_tile = [&](int kt, unsigned short* A, unsigned short* B) {
#pragma unroll
    for (int i = 0; i < 2; i++) {
      int c = tid + 256 * i;
      int r = c >> 2, col = (c & 3) * 8;
      const unsigned short* src = ab + (size_t)(rowbase + r) * DIM + kt + col;
      unsigned short* dst = A + r * 40 + col;
#if HAVE_ASYNC
      async_copy_b128(src, dst);
#else
      *(u32x4*)dst = *(const u32x4*)src;
#endif
    }
    {
      int kp = (tid & 15) * 2;
      int nc = (tid >> 4) * 8;
      const unsigned short* s0 = wb + (size_t)(kt + kp) * DIM + colbase + nc;
      u32x4 d0 = *(const u32x4*)s0;
      u32x4 d1 = *(const u32x4*)(s0 + DIM);
#pragma unroll
      for (int e = 0; e < 8; e++) {
        unsigned lo = (d0[e >> 1] >> (16 * (e & 1))) & 0xffffu;
        unsigned hi = (d1[e >> 1] >> (16 * (e & 1))) & 0xffffu;
        *(unsigned*)(B + (nc + e) * 40 + kp) = lo | (hi << 16);
      }
    }
  };

  load_tile(0, smem, smem + 10240);
  wait_async0();
  __syncthreads();

  for (int t = 0; t < DIM / 32; ++t) {
    unsigned short* Ac = smem + (t & 1) * 5120;
    unsigned short* Bc = smem + 10240 + (t & 1) * 5120;
    if (t + 1 < DIM / 32)
      load_tile((t + 1) * 32, smem + ((t + 1) & 1) * 5120,
                smem + 10240 + ((t + 1) & 1) * 5120);

    v16bf af[2], bfm[4];
#pragma unroll
    for (int i = 0; i < 2; i++) af[i] = frag_a(Ac + (32 * wm + 16 * i) * 40, 40);
#pragma unroll
    for (int j = 0; j < 4; j++) bfm[j] = frag_b(Bc + (64 * wn + 16 * j) * 40, 40);
#pragma unroll
    for (int i = 0; i < 2; i++)
#pragma unroll
      for (int j = 0; j < 4; j++) acc[i][j] = wmma_bf16(af[i], bfm[j], acc[i][j]);

    wait_async0();
    __syncthreads();
  }

#pragma unroll
  for (int i = 0; i < 2; i++)
#pragma unroll
    for (int j = 0; j < 4; j++)
#pragma unroll
      for (int r = 0; r < 8; r++) {
        int row = rowbase + 32 * wm + 16 * i + r + 8 * hf;
        int col = colbase + 64 * wn + 16 * j + ln;
        out[(size_t)row * DIM + col] = acc[i][j][r] + bout[col];
      }
}

// ---------------------------------------------------------------------------
extern "C" void kernel_launch(void* const* d_in, const int* in_sizes, int n_in,
                              void* d_out, int out_size, void* d_ws, size_t ws_size,
                              hipStream_t stream) {
  const float* x    = (const float*)d_in[0];
  const float* Wqkv = (const float*)d_in[1];
  const float* bqkv = (const float*)d_in[2];
  const float* Wout = (const float*)d_in[3];
  const float* bout = (const float*)d_in[4];
  float* out = (float*)d_out;

  const size_t NX = (size_t)BATCH * SEQ * DIM;            // 8,388,608
  unsigned short* xb    = (unsigned short*)d_ws;
  unsigned short* wqkvb = xb + NX;
  unsigned short* woutb = wqkvb + (size_t)DIM * 3 * DIM;
  unsigned short* qb    = woutb + (size_t)DIM * DIM;
  unsigned short* kb    = qb + NX;
  unsigned short* vb    = kb + NX;
  unsigned short* abuf  = vb + NX;

  cvt_bf16<<<(int)(NX / 1024), 256, 0, stream>>>(x, xb, (int)NX);
  cvt_bf16<<<(DIM * 3 * DIM) / 1024, 256, 0, stream>>>(Wqkv, wqkvb, DIM * 3 * DIM);
  cvt_bf16<<<(DIM * DIM) / 1024, 256, 0, stream>>>(Wout, woutb, DIM * DIM);

  qkv_gemm_rope<<<dim3(3 * DIM / 128, BATCH * SEQ / 128), 256, 0, stream>>>(
      xb, wqkvb, bqkv, qb, kb, vb);

  attn_kernel<<<dim3(SEQ / 128, BATCH * NHEAD), 256, 0, stream>>>(qb, kb, vb, abuf);

  out_gemm<<<dim3(DIM / 128, BATCH * SEQ / 128), 256, 0, stream>>>(
      abuf, woutb, bout, out);
}
